// CustomGNNLayer3_91319594647968
// MI455X (gfx1250) — compile-verified, matrix operand order checked by reading, not run on hardware
//
#include <hip/hip_runtime.h>
#include <cstdint>
#include <cstddef>

// ---- problem constants ----
#define N_ROWS 8192
#define F_DIM  256
#define C1 0.9f            // 1 - 2*RHO*ALPHA
#define C4 0.1f            // 2*RHO*ALPHA
#define RHOBETA 0.005f     // RHO*BETA
#define BN_EPS 1e-5f

typedef __attribute__((ext_vector_type(16))) __bf16 v16bf;
typedef __attribute__((ext_vector_type(8)))  __bf16 v8bf;
typedef __attribute__((ext_vector_type(8)))  float  v8f;

// ---- WMMA helper (CDNA5 16x16x32 bf16) ----
__device__ __forceinline__ v8f wmma_bf16(v16bf a, v16bf b, v8f c) {
  return __builtin_amdgcn_wmma_f32_16x16x32_bf16(
      false, a, false, b, (short)0, c, false, false);
}

// ---- CDNA5 async Global->LDS copy (ASYNCcnt-tracked, bypasses VGPRs) ----
__device__ __forceinline__ void async_copy16(uint32_t lds_off, const void* g) {
  asm volatile("global_load_async_to_lds_b128 %0, %1, off"
               :: "v"(lds_off), "v"((uint64_t)(uintptr_t)g)
               : "memory");
}
__device__ __forceinline__ void wait_async_le4() {
  asm volatile("s_wait_asynccnt 0x4" ::: "memory");
}
__device__ __forceinline__ void wait_async_le0() {
  asm volatile("s_wait_asynccnt 0x0" ::: "memory");
}

// A-fragment (also valid as B-fragment when B = srcT, i.e. B[k][n] = src[n][k]):
// lane L: row = row0 + (L&15); K elems = {kb..kb+7, kb+16..kb+23}, kb = k0 + 8*(L>=16)
__device__ __forceinline__ v16bf load_frag(const __bf16* p, int row0, int k0, int ld) {
  const int l  = threadIdx.x & 31;
  const int r  = row0 + (l & 15);
  const int kb = k0 + ((l >> 4) << 3);
  const __bf16* q = p + (size_t)r * ld + kb;
  v8bf lo = *(const v8bf*)(q);
  v8bf hi = *(const v8bf*)(q + 16);
  v16bf f;
#pragma unroll
  for (int i = 0; i < 8; ++i) { f[i] = lo[i]; f[i + 8] = hi[i]; }
  return f;
}

// ---------------- prep: bf16 casts, transpose, Y = c1*H + c4*X, r = ||H_n||^2 ----
__global__ __launch_bounds__(256) void prep_kernel(
    const float* __restrict__ H, const float* __restrict__ X, const float* __restrict__ W,
    __bf16* __restrict__ Hb, __bf16* __restrict__ HbT, __bf16* __restrict__ Yb,
    __bf16* __restrict__ Wb, float* __restrict__ rv) {
  const int n = blockIdx.x, s = threadIdx.x;
  const size_t idx = (size_t)n * F_DIM + s;
  const float h = H[idx];
  Hb[idx] = (__bf16)h;
  HbT[(size_t)s * N_ROWS + n] = (__bf16)h;
  Yb[idx] = (__bf16)(C1 * h + C4 * X[idx]);
  if (n < F_DIM) Wb[idx] = (__bf16)W[idx];
  __shared__ float sh[256];
  sh[s] = h * h;
  __syncthreads();
  for (int off = 128; off > 0; off >>= 1) {
    if (s < off) sh[s] += sh[s + off];
    __syncthreads();
  }
  if (s == 0) rv[n] = sh[0];
}

// ---------------- gram_exp: E = f(H H^T), Zp = partial rowsum(E) ----------------
// grid (N/128, 2): blockIdx.y selects column half; async double-buffered staging.
__global__ __launch_bounds__(256) void gram_exp_kernel(
    const __bf16* __restrict__ Hb, const float* __restrict__ rv,
    __bf16* __restrict__ E, float* __restrict__ Zp) {
  __shared__ __align__(16) __bf16 sHj[2][32 * F_DIM];   // 2 x 16 KB
  const int wave = threadIdx.x >> 5, lane = threadIdx.x & 31;
  const int m0 = blockIdx.x * 128 + wave * 16;
  const int mb = m0 + ((lane >> 4) << 3);

  v16bf afr[8];
#pragma unroll
  for (int k = 0; k < 8; ++k) afr[k] = load_frag(Hb, m0, k * 32, F_DIM);
  float rA[8];
#pragma unroll
  for (int v = 0; v < 8; ++v) rA[v] = rv[mb + v];
  float zacc[8];
#pragma unroll
  for (int v = 0; v < 8; ++v) zacc[v] = 0.f;
  const float invTwo = 1.0f / (2.0f + 1e-7f);   // SCALE=1 -> 1/(2*sigma^2+1e-7)

  auto stage = [&](int j0, int buf) {
    const __bf16* src = Hb + (size_t)j0 * F_DIM + threadIdx.x * 8;
    const uint32_t dst = (uint32_t)(uintptr_t)&sHj[buf][0] + threadIdx.x * 16;
#pragma unroll
    for (int i = 0; i < 4; ++i) async_copy16(dst + i * 4096, src + i * 2048);
  };

  const int jbeg = blockIdx.y * (N_ROWS / 2);
  const int NB = (N_ROWS / 2) / 32;             // 128 col blocks per half
  stage(jbeg, 0);
  for (int jb = 0; jb < NB; ++jb) {
    const int j0 = jbeg + jb * 32;
    if (jb + 1 < NB) { stage(j0 + 32, (jb + 1) & 1); wait_async_le4(); }
    else             { wait_async_le0(); }
    __syncthreads();
    const __bf16* cur = &sHj[jb & 1][0];
#pragma unroll
    for (int t = 0; t < 2; ++t) {
      v16bf bfr[8];
#pragma unroll
      for (int k = 0; k < 8; ++k) bfr[k] = load_frag(cur, t * 16, k * 32, F_DIM);
      v8f acc;
#pragma unroll
      for (int i = 0; i < 8; ++i) acc[i] = 0.f;
#pragma unroll
      for (int k = 0; k < 8; ++k) acc = wmma_bf16(afr[k], bfr[k], acc);

      const int jcol = j0 + t * 16 + (lane & 15);
      const float rB = rv[jcol];
#pragma unroll
      for (int v = 0; v < 8; ++v) {
        float d  = rA[v] + rB - 2.0f * acc[v];
        d        = fmaxf(d, 0.0f);
        float ds = fminf(d * invTwo, 100.0f);
        float ev = fmaxf(__expf(-ds), 1e-7f);
        zacc[v] += ev;
        E[(size_t)(mb + v) * N_ROWS + jcol] = (__bf16)ev;
      }
    }
    __syncthreads();
  }
#pragma unroll
  for (int v = 0; v < 8; ++v) {
#pragma unroll
    for (int o = 1; o < 16; o <<= 1) zacc[v] += __shfl_xor(zacc[v], o, 32);
  }
  if ((lane & 15) == 0) {
#pragma unroll
    for (int v = 0; v < 8; ++v) Zp[(size_t)blockIdx.y * N_ROWS + mb + v] = zacc[v];
  }
}

// ---------------- pe_gemm: Pu = E @ H (K = 8192) --------------------------------
// grid 128: block = 64 rows; wave w: rows (w&3)*16, col tiles [(w>>2)*8, +8).
__global__ __launch_bounds__(256) void pe_gemm_kernel(
    const __bf16* __restrict__ E, const __bf16* __restrict__ HbT, float* __restrict__ Pu) {
  __shared__ __align__(16) __bf16 sB[2][F_DIM * 32];   // 2 x 16 KB
  const int wave = threadIdx.x >> 5, lane = threadIdx.x & 31;
  const int m0 = blockIdx.x * 64 + (wave & 3) * 16;
  const int ntb = (wave >> 2) * 8;
  v8f acc[8];
#pragma unroll
  for (int nt = 0; nt < 8; ++nt)
#pragma unroll
    for (int i = 0; i < 8; ++i) acc[nt][i] = 0.f;

  auto stageB = [&](int k0, int buf) {
    const __bf16* src = HbT + (size_t)threadIdx.x * N_ROWS + k0;
    const uint32_t dst = (uint32_t)(uintptr_t)&sB[buf][0] + threadIdx.x * 64;
#pragma unroll
    for (int i = 0; i < 4; ++i) async_copy16(dst + i * 16, src + i * 8);
  };

  const int NB = N_ROWS / 32;                   // 256 K blocks
  stageB(0, 0);
  for (int kb = 0; kb < NB; ++kb) {
    if (kb + 1 < NB) { stageB((kb + 1) * 32, (kb + 1) & 1); wait_async_le4(); }
    else             { wait_async_le0(); }
    __syncthreads();
    const __bf16* cur = &sB[kb & 1][0];
    v16bf a = load_frag(E, m0, kb * 32, N_ROWS);
#pragma unroll
    for (int nt = 0; nt < 8; ++nt) {
      v16bf bfr = load_frag(cur, (ntb + nt) * 16, 0, 32);
      acc[nt] = wmma_bf16(a, bfr, acc[nt]);
    }
    __syncthreads();
  }
  const int mb = m0 + ((lane >> 4) << 3);
#pragma unroll
  for (int nt = 0; nt < 8; ++nt) {
    const int col = (ntb + nt) * 16 + (lane & 15);
#pragma unroll
    for (int v = 0; v < 8; ++v) Pu[(size_t)(mb + v) * F_DIM + col] = acc[nt][v];
  }
}

// ---------------- rowstats: pp = sum(P^2), g = sum(H*P)/Z, P = Pu/Z -------------
__global__ __launch_bounds__(256) void rowstats_kernel(
    const float* __restrict__ Pu, const float* __restrict__ H, const float* __restrict__ Zp,
    float* __restrict__ pp, float* __restrict__ g) {
  const int n = blockIdx.x, s = threadIdx.x;
  const float z = Zp[n] + Zp[N_ROWS + n];
  const float p = Pu[(size_t)n * F_DIM + s] / z;
  const float h = H[(size_t)n * F_DIM + s];
  __shared__ float s1[256], s2[256];
  s1[s] = p * p; s2[s] = h * p;
  __syncthreads();
  for (int off = 128; off > 0; off >>= 1) {
    if (s < off) { s1[s] += s1[s + off]; s2[s] += s2[s + off]; }
    __syncthreads();
  }
  if (s == 0) { pp[n] = s1[0]; g[n] = s2[0] / z; }
}

// ---------------- etg_part: u_partial = E^T g over 128-row chunks ---------------
__global__ __launch_bounds__(256) void etg_part_kernel(
    const __bf16* __restrict__ E, const float* __restrict__ g, float* __restrict__ upart) {
  const int cb = blockIdx.x & 31;        // 32 column blocks of 256
  const int ch = blockIdx.x >> 5;        // 64 row chunks of 128
  const int col = cb * 256 + threadIdx.x;
  const int n0 = ch * 128;
  __shared__ float sg[128];
  if (threadIdx.x < 128) sg[threadIdx.x] = g[n0 + threadIdx.x];
  __syncthreads();
  float acc = 0.f;
  for (int i = 0; i < 128; ++i)
    acc += (float)E[(size_t)(n0 + i) * N_ROWS + col] * sg[i];
  upart[(size_t)ch * N_ROWS + col] = acc;
}

// ---------------- tv = RHOBETA * (pp + 2 * sum_ch upart) ------------------------
__global__ __launch_bounds__(256) void tv_kernel(
    const float* __restrict__ upart, const float* __restrict__ pp, float* __restrict__ tv) {
  const int n = blockIdx.x * 256 + threadIdx.x;
  float u = 0.f;
  for (int ch = 0; ch < 64; ++ch) u += upart[(size_t)ch * N_ROWS + n];
  tv[n] = RHOBETA * (pp[n] + 2.0f * u);
}

// ---------------- norm of tv ----------------------------------------------------
__global__ __launch_bounds__(256) void norm_kernel(const float* __restrict__ tv,
                                                   float* __restrict__ nrm) {
  const int t = threadIdx.x;
  float s = 0.f;
  for (int i = t; i < N_ROWS; i += 256) { float v = tv[i]; s += v * v; }
  __shared__ float sh[256];
  sh[t] = s; __syncthreads();
  for (int off = 128; off > 0; off >>= 1) {
    if (t < off) sh[t] += sh[t + off];
    __syncthreads();
  }
  if (t == 0) nrm[0] = sqrtf(sh[0]);
}

// ---------------- out_gemm: O = Y W^T + b + tv/nrm ------------------------------
// grid 128: block = 64 rows; wave w: rows (w&3)*16, col tiles [(w>>2)*8, +8).
__global__ __launch_bounds__(256) void out_gemm_kernel(
    const __bf16* __restrict__ Yb, const __bf16* __restrict__ Wb,
    const float* __restrict__ bvec, const float* __restrict__ tv,
    const float* __restrict__ nrmp, float* __restrict__ O) {
  const int wave = threadIdx.x >> 5, lane = threadIdx.x & 31;
  const int m0 = blockIdx.x * 64 + (wave & 3) * 16;
  const int ntb = (wave >> 2) * 8;
  const int mb = m0 + ((lane >> 4) << 3);
  v16bf a[8];
#pragma unroll
  for (int k = 0; k < 8; ++k) a[k] = load_frag(Yb, m0, k * 32, F_DIM);
  float nrm = nrmp[0];
  if (nrm == 0.f) nrm = 1.f;
  float t5[8];
#pragma unroll
  for (int v = 0; v < 8; ++v) t5[v] = tv[mb + v] / nrm;
#pragma unroll
  for (int nt = 0; nt < 8; ++nt) {
    v8f acc;
#pragma unroll
    for (int i = 0; i < 8; ++i) acc[i] = 0.f;
#pragma unroll
    for (int k = 0; k < 8; ++k) {
      v16bf b = load_frag(Wb, (ntb + nt) * 16, k * 32, F_DIM);   // B = W^T frag
      acc = wmma_bf16(a[k], b, acc);
    }
    const int col = (ntb + nt) * 16 + (lane & 15);
    const float bb = bvec[col];
#pragma unroll
    for (int v = 0; v < 8; ++v)
      O[(size_t)(mb + v) * F_DIM + col] = acc[v] + bb + t5[v];
  }
}

// ---------------- BN column stats (deterministic two-stage) ---------------------
__global__ __launch_bounds__(256) void colstats_part_kernel(
    const float* __restrict__ O, float* __restrict__ ps, float* __restrict__ ps2) {
  const int cg = blockIdx.x & 7;       // 8 groups of 32 columns
  const int rc = blockIdx.x >> 3;      // 8 chunks of 1024 rows
  const int t = threadIdx.x;
  const int col = cg * 32 + (t & 31);
  float s = 0.f, s2 = 0.f;
  for (int r = rc * 1024 + (t >> 5); r < rc * 1024 + 1024; r += 8) {
    float v = O[(size_t)r * F_DIM + col];
    s += v; s2 += v * v;
  }
  __shared__ float sh1[256], sh2[256];
  sh1[t] = s; sh2[t] = s2;
  __syncthreads();
  for (int off = 128; off >= 32; off >>= 1) {
    if (t < off) { sh1[t] += sh1[t + off]; sh2[t] += sh2[t + off]; }
    __syncthreads();
  }
  if (t < 32) { ps[rc * F_DIM + col] = sh1[t]; ps2[rc * F_DIM + col] = sh2[t]; }
}

__global__ void colstats_fin_kernel(const float* __restrict__ ps, const float* __restrict__ ps2,
                                    float* __restrict__ mu, float* __restrict__ invstd) {
  const int c = threadIdx.x;
  float s = 0.f, s2 = 0.f;
  for (int rc = 0; rc < 8; ++rc) { s += ps[rc * F_DIM + c]; s2 += ps2[rc * F_DIM + c]; }
  const float m = s / (float)N_ROWS;
  const float var = s2 / (float)N_ROWS - m * m;
  mu[c] = m;
  invstd[c] = rsqrtf(var + BN_EPS);
}

__global__ __launch_bounds__(256) void bn_final_kernel(
    const float* __restrict__ O, const float* __restrict__ mu, const float* __restrict__ invstd,
    const float* __restrict__ gamma, const float* __restrict__ beta, float* __restrict__ out) {
  const int n = blockIdx.x, c = threadIdx.x;
  const size_t idx = (size_t)n * F_DIM + c;
  out[idx] = (O[idx] - mu[c]) * invstd[c] * gamma[c] + beta[c];
}

// ---------------- launcher ------------------------------------------------------
extern "C" void kernel_launch(void* const* d_in, const int* in_sizes, int n_in,
                              void* d_out, int out_size, void* d_ws, size_t ws_size,
                              hipStream_t stream) {
  (void)in_sizes; (void)n_in; (void)out_size; (void)ws_size;
  const float* H     = (const float*)d_in[0];
  const float* X     = (const float*)d_in[1];
  const float* W     = (const float*)d_in[2];
  const float* bvec  = (const float*)d_in[3];
  const float* gamma = (const float*)d_in[4];
  const float* beta  = (const float*)d_in[5];
  float* out = (float*)d_out;

  char* base = (char*)d_ws;
  size_t off = 0;
  auto carve = [&](size_t bytes) -> char* {
    char* q = base + off;
    off = (off + bytes + 255) & ~(size_t)255;
    return q;
  };
  __bf16* E    = (__bf16*)carve((size_t)N_ROWS * N_ROWS * 2);   // 128 MB, L2-resident
  __bf16* Hb   = (__bf16*)carve((size_t)N_ROWS * F_DIM * 2);
  __bf16* HbT  = (__bf16*)carve((size_t)F_DIM * N_ROWS * 2);
  __bf16* Yb   = (__bf16*)carve((size_t)N_ROWS * F_DIM * 2);
  __bf16* Wb   = (__bf16*)carve((size_t)F_DIM * F_DIM * 2);
  float*  rv   = (float*)carve((size_t)N_ROWS * 4);
  float*  Zp   = (float*)carve((size_t)2 * N_ROWS * 4);
  float*  Pu   = (float*)carve((size_t)N_ROWS * F_DIM * 4);
  float*  pp   = (float*)carve((size_t)N_ROWS * 4);
  float*  g    = (float*)carve((size_t)N_ROWS * 4);
  float*  up   = (float*)carve((size_t)64 * N_ROWS * 4);
  float*  tv   = (float*)carve((size_t)N_ROWS * 4);
  float*  nrm  = (float*)carve(256);
  float*  O    = (float*)carve((size_t)N_ROWS * F_DIM * 4);
  float*  ps   = (float*)carve((size_t)8 * F_DIM * 4);
  float*  ps2  = (float*)carve((size_t)8 * F_DIM * 4);
  float*  mu   = (float*)carve((size_t)F_DIM * 4);
  float*  istd = (float*)carve((size_t)F_DIM * 4);

  prep_kernel<<<N_ROWS, 256, 0, stream>>>(H, X, W, Hb, HbT, Yb, Wb, rv);
  gram_exp_kernel<<<dim3(N_ROWS / 128, 2), 256, 0, stream>>>(Hb, rv, E, Zp);
  pe_gemm_kernel<<<N_ROWS / 64, 256, 0, stream>>>(E, HbT, Pu);
  rowstats_kernel<<<N_ROWS, 256, 0, stream>>>(Pu, H, Zp, pp, g);
  etg_part_kernel<<<32 * 64, 256, 0, stream>>>(E, g, up);
  tv_kernel<<<N_ROWS / 256, 256, 0, stream>>>(up, pp, tv);
  norm_kernel<<<1, 256, 0, stream>>>(tv, nrm);
  out_gemm_kernel<<<N_ROWS / 64, 256, 0, stream>>>(Yb, Wb, bvec, tv, nrm, O);
  colstats_part_kernel<<<64, 256, 0, stream>>>(O, ps, ps2);
  colstats_fin_kernel<<<1, 256, 0, stream>>>(ps, ps2, mu, istd);
  bn_final_kernel<<<N_ROWS, 256, 0, stream>>>(O, mu, istd, gamma, beta, out);
}
